// MultiHeadAttention_7069516169541
// MI455X (gfx1250) — matrix operand order repaired and checked by reading, not compiled, not run on hardware
//
#include <hip/hip_runtime.h>
#include <stdint.h>

// ---------------------------------------------------------------------------
// MHA forward for B=2, S=2048, D=768, H=12, hd=64 on gfx1250 (CDNA5, wave32).
// Structure: one-shot f32->bf16 conversion pass, then all GEMM tiles staged by
// pure async global->LDS DMA (double-buffered, s_wait_asynccnt ping-pong),
// bf16 WMMA with f32 accumulate, flash-attention inner loop.
// Main loops forced to unroll(1) so accumulators coalesce in-place (no
// v_mov_b64 churn / WMMA->VALU hazard NOPs between WMMAs).
// ---------------------------------------------------------------------------

#define SEQ    2048
#define DMODEL 768
#define HEADS  12
#define HD     64

typedef __attribute__((ext_vector_type(16))) __bf16 v16bf;
typedef __attribute__((ext_vector_type(8)))  __bf16 v8bf;
typedef __attribute__((ext_vector_type(8)))  float  v8f;

// f32 -> packed bf16 via hardware converter (RNE)
__device__ __forceinline__ uint32_t pk2(float a, float b) {
    uint32_t r;
    asm("v_cvt_pk_bf16_f32 %0, %1, %2" : "=v"(r) : "v"(a), "v"(b));
    return r;
}
__device__ __forceinline__ unsigned short f2bf(float f) {
    uint32_t r;
    asm("v_cvt_pk_bf16_f32 %0, %1, %1" : "=v"(r) : "v"(f));
    return (unsigned short)r;
}

// async global->LDS 16B per lane (ASYNCcnt); INST_OFFSET applies to both sides
__device__ __forceinline__ void async_b128(uint32_t lds_off, uint64_t gaddr) {
    asm volatile("global_load_async_to_lds_b128 %0, %1, off"
                 :: "v"(lds_off), "v"(gaddr) : "memory");
}
__device__ __forceinline__ void async_b128_o16(uint32_t lds_off, uint64_t gaddr) {
    asm volatile("global_load_async_to_lds_b128 %0, %1, off offset:16"
                 :: "v"(lds_off), "v"(gaddr) : "memory");
}
__device__ __forceinline__ void async_b128_o32(uint32_t lds_off, uint64_t gaddr) {
    asm volatile("global_load_async_to_lds_b128 %0, %1, off offset:32"
                 :: "v"(lds_off), "v"(gaddr) : "memory");
}
__device__ __forceinline__ void async_b128_o48(uint32_t lds_off, uint64_t gaddr) {
    asm volatile("global_load_async_to_lds_b128 %0, %1, off offset:48"
                 :: "v"(lds_off), "v"(gaddr) : "memory");
}
__device__ __forceinline__ void wait_async0() {
    asm volatile("s_wait_asynccnt 0" ::: "memory");
}
__device__ __forceinline__ void wait_async2() {
    asm volatile("s_wait_asynccnt 2" ::: "memory");
}
__device__ __forceinline__ void wait_async6() {
    asm volatile("s_wait_asynccnt 6" ::: "memory");
}
__device__ __forceinline__ uint32_t lds_off32(const void* p) {
    return (uint32_t)(uintptr_t)p;   // generic LDS pointer: low 32 bits = LDS addr
}

// cross-lane xor exchange within 16-lane groups (DPP / ds_swizzle)
__device__ __forceinline__ float lane_xor1(float x) {   // quad_perm(1,0,3,2)
    return __builtin_bit_cast(float,
        __builtin_amdgcn_mov_dpp(__builtin_bit_cast(int, x), 0xB1, 0xF, 0xF, true));
}
__device__ __forceinline__ float lane_xor2(float x) {   // quad_perm(2,3,0,1)
    return __builtin_bit_cast(float,
        __builtin_amdgcn_mov_dpp(__builtin_bit_cast(int, x), 0x4E, 0xF, 0xF, true));
}
__device__ __forceinline__ float lane_xor4(float x) {   // ds_swizzle SWAPX4
    return __builtin_bit_cast(float,
        __builtin_amdgcn_ds_swizzle(__builtin_bit_cast(int, x), 0x101F));
}
__device__ __forceinline__ float lane_xor8(float x) {   // row_ror:8 == xor 8 (mod 16)
    return __builtin_bit_cast(float,
        __builtin_amdgcn_mov_dpp(__builtin_bit_cast(int, x), 0x128, 0xF, 0xF, true));
}
__device__ __forceinline__ float red_max16(float x) {
    x = fmaxf(x, lane_xor1(x)); x = fmaxf(x, lane_xor2(x));
    x = fmaxf(x, lane_xor4(x)); x = fmaxf(x, lane_xor8(x));
    return x;
}
__device__ __forceinline__ float red_sum16(float x) {
    x += lane_xor1(x); x += lane_xor2(x); x += lane_xor4(x); x += lane_xor8(x);
    return x;
}

__device__ __forceinline__ v16bf cat8(v8bf lo, v8bf hi) {
    return __builtin_shufflevector(lo, hi, 0,1,2,3,4,5,6,7,8,9,10,11,12,13,14,15);
}
// A-fragment (16x32 bf16, M x K): lane holds row m=lane&15, half=lane>>4.
__device__ __forceinline__ v16bf load_afrag(const unsigned short* row, int half) {
    v8bf lo = *(const v8bf*)(row + 16 * half);
    v8bf hi = *(const v8bf*)(row + 16 * half + 8);
    return cat8(lo, hi);
}
// B-fragment (32x16 bf16, K x N): lane holds col n=lane&15; 16 contiguous k.
__device__ __forceinline__ v16bf load_bfrag(const unsigned short* row, int half) {
    return *(const v16bf*)(row + 16 * half);
}
__device__ __forceinline__ v8f wmma_bf16(v16bf a, v16bf b, v8f c) {
    return __builtin_amdgcn_wmma_f32_16x16x32_bf16(false, a, false, b,
                                                   (short)0, c, false, false);
}

// ===========================================================================
// Kernel 0: one-shot f32 -> bf16 conversion (8 elems/thread, b128 in/out)
// ===========================================================================
__global__ __launch_bounds__(256)
void cvt_f32_bf16_kernel(const float* __restrict__ in,
                         unsigned short* __restrict__ out, int n8)
{
    int i = blockIdx.x * 256 + threadIdx.x;
    if (i < n8) {
        const float4* s = (const float4*)(in + (size_t)i * 8);
        float4 a = s[0], b = s[1];
        *(uint4*)(out + (size_t)i * 8) =
            make_uint4(pk2(a.x,a.y), pk2(a.z,a.w), pk2(b.x,b.y), pk2(b.z,b.w));
    }
}

// ===========================================================================
// Kernel 1: projection  Out = rope?( Xbf @ Wbf^T + bias )  bf16 [B,H,S,64]
// 256 thr / 8 waves; tile M=128 x N=64, K-step 64 (8 WMMA per barrier).
// A+B tiles staged by async DMA, double-buffered (6 asyncs/tile/wave).
// ===========================================================================
__global__ __launch_bounds__(256)
void proj_rope_kernel(const unsigned short* __restrict__ X,  // [4096][768] bf16
                      const unsigned short* __restrict__ W,  // [768][768]  bf16
                      const float* __restrict__ bias,
                      unsigned short* __restrict__ Out,      // [B][H][S][64] bf16
                      int applyRope)
{
    __shared__ __attribute__((aligned(32))) unsigned short lA[2][128][80]; // 64 used
    __shared__ __attribute__((aligned(32))) unsigned short lB[2][64][80];

    const int tid   = threadIdx.x;
    const int lane  = tid & 31;
    const int wave  = tid >> 5;
    const int waveM = wave & 3;
    const int waveN = wave >> 2;
    const int blockM = (blockIdx.x & 31) * 128;
    const int blockN = (blockIdx.x >> 5) * 64;
    const int mrow = lane & 15;
    const int half = lane >> 4;

    // loader mapping
    const int aR = tid >> 1, aK = (tid & 1) * 32;       // A: 128 rows x 32 elems
    const int bN = tid >> 2, bK = (tid & 3) * 16;       // B: 64 rows x 16 elems
    const unsigned short* Ald = X + (size_t)(blockM + aR) * DMODEL + aK;
    const unsigned short* Bld = W + (size_t)(blockN + bN) * DMODEL + bK;

    v8f acc[2][2] = {};
    const int nK = DMODEL / 64;   // 12

    // prologue: stage k-tile 0 into buffer 0
    {
        uint64_t ga = (uint64_t)(uintptr_t)Ald;
        uint32_t la = lds_off32(&lA[0][aR][aK]);
        async_b128(la, ga); async_b128_o16(la, ga);
        async_b128_o32(la, ga); async_b128_o48(la, ga);
        uint64_t gb = (uint64_t)(uintptr_t)Bld;
        uint32_t lb = lds_off32(&lB[0][bN][bK]);
        async_b128(lb, gb); async_b128_o16(lb, gb);
    }

    #pragma unroll 1    // keep single body so accumulators coalesce in-place
    for (int t = 0; t < nK; ++t) {
        const int buf = t & 1;
        if (t + 1 < nK) {      // stage next k-tile into other buffer
            int k0n = (t + 1) * 64;
            uint64_t ga = (uint64_t)(uintptr_t)(Ald + k0n);
            uint32_t la = lds_off32(&lA[buf ^ 1][aR][aK]);
            async_b128(la, ga); async_b128_o16(la, ga);
            async_b128_o32(la, ga); async_b128_o48(la, ga);
            uint64_t gb = (uint64_t)(uintptr_t)(Bld + k0n);
            uint32_t lb = lds_off32(&lB[buf ^ 1][bN][bK]);
            async_b128(lb, gb); async_b128_o16(lb, gb);
            wait_async6();     // in-order: tile t's 6 DMAs complete
        } else {
            wait_async0();
        }
        __syncthreads();
        #pragma unroll
        for (int ks = 0; ks < 2; ++ks) {
            v16bf a0 = load_afrag(&lA[buf][waveM*32      + mrow][ks*32], half);
            v16bf a1 = load_afrag(&lA[buf][waveM*32 + 16 + mrow][ks*32], half);
            v16bf b0 = load_bfrag(&lB[buf][waveN*32      + mrow][ks*32], half);
            v16bf b1 = load_bfrag(&lB[buf][waveN*32 + 16 + mrow][ks*32], half);
            acc[0][0] = wmma_bf16(a0, b0, acc[0][0]);
            acc[0][1] = wmma_bf16(a0, b1, acc[0][1]);
            acc[1][0] = wmma_bf16(a1, b0, acc[1][0]);
            acc[1][1] = wmma_bf16(a1, b1, acc[1][1]);
        }
        __syncthreads();
    }

    // ---- epilogue: bias + RoPE + scatter to [B][H][S][64] bf16 ----
    #pragma unroll
    for (int mi = 0; mi < 2; ++mi)
    #pragma unroll
    for (int ni = 0; ni < 2; ++ni) {
        int gn   = blockN + waveN*32 + ni*16 + mrow;
        float bv = bias[gn];
        int head = gn >> 6;
        int hdim = gn & 63;
        int doRot = applyRope && (hdim < 32);          // uniform across fragment
        int p = hdim >> 1;
        float inv_freq = exp2f((float)p * -0.8304820237f); // 10000^(-p/16)
        float sgn = (hdim & 1) ? 1.0f : -1.0f;
        #pragma unroll
        for (int r = 0; r < 8; ++r) {
            int gm = blockM + waveM*32 + mi*16 + r + 8*half;
            int b  = gm >> 11;
            int s  = gm & (SEQ - 1);
            float v = acc[mi][ni][r] + bv;
            float partner = lane_xor1(v);              // dims 2p <-> 2p+1 are lane^1
            if (doRot) {
                float ang = (float)s * inv_freq;
                float sn, cs;
                __sincosf(ang, &sn, &cs);
                v = v * cs + sgn * partner * sn;
            }
            Out[(((size_t)b * HEADS + head) * SEQ + s) * HD + hdim] = f2bf(v);
        }
    }
}

// ===========================================================================
// Kernel 2: flash attention, causal, double-buffered KV staging.
// K tile: async DMA ping-pong; V tile: global->regs prefetch + LDS transpose.
// ===========================================================================
__global__ __launch_bounds__(256)
void flash_attn_kernel(const unsigned short* __restrict__ Q,
                       const unsigned short* __restrict__ K,
                       const unsigned short* __restrict__ V,
                       unsigned short* __restrict__ Ctx)
{
    __shared__ __attribute__((aligned(32))) unsigned short lK[2][64][80];  // [buf][kv][hd]
    __shared__ __attribute__((aligned(32))) unsigned short lV[2][64][80];  // [buf][hd][kv]
    __shared__ __attribute__((aligned(32))) unsigned short lP[8][16][64];  // per-wave P

    const int tid   = threadIdx.x;
    const int lane  = tid & 31;
    const int wave  = tid >> 5;
    const int qTile = blockIdx.x & 15;
    const int bh    = blockIdx.x >> 4;
    const int qBase = qTile * 128;
    const size_t base = (size_t)bh * SEQ * HD;
    const int mrow = lane & 15;
    const int half = lane >> 4;

    const int ldR   = tid >> 2;
    const int ldSeg = (tid & 3) * 16;
    const unsigned short* Kld = K + base + ldSeg;
    const unsigned short* Vld = V + base + ldSeg;

    // Q A-fragments from global (row-major [s][hd] == A layout)
    v16bf qa[2];
    {
        const unsigned short* qp = Q + base + (size_t)(qBase + wave*16 + mrow) * HD;
        #pragma unroll
        for (int ks = 0; ks < 2; ++ks) {
            v8bf lo = *(const v8bf*)(qp + ks*32 + 16*half);
            v8bf hi = *(const v8bf*)(qp + ks*32 + 16*half + 8);
            qa[ks] = cat8(lo, hi);
        }
    }

    v8f o[4] = {};
    float mi[8], li[8];
    #pragma unroll
    for (int r = 0; r < 8; ++r) { mi[r] = -3.0e38f; li[r] = 0.0f; }

    const float scale = 0.125f;
    const int nTiles = qTile * 2 + 2;    // always >= 2

    uint32_t vreg[8];
    {   // prologue: stage tile 0
        uint64_t ga = (uint64_t)(uintptr_t)(Kld + (size_t)ldR * HD);
        uint32_t la = lds_off32(&lK[0][ldR][ldSeg]);
        async_b128(la, ga); async_b128_o16(la, ga);
        const uint4* vp = (const uint4*)(Vld + (size_t)ldR * HD);
        *(uint4*)&vreg[0] = vp[0];
        *(uint4*)&vreg[4] = vp[1];
    }

    #pragma unroll 1    // keep single body so O-accumulators coalesce in-place
    for (int t = 0; t < nTiles; ++t) {
        const int kv0 = t * 64;
        const int buf = t & 1;

        // V regs (tile t) -> lV[buf] transposed
        #pragma unroll
        for (int i = 0; i < 8; ++i) {
            lV[buf][ldSeg + 2*i][ldR]     = (unsigned short)(vreg[i]);
            lV[buf][ldSeg + 2*i + 1][ldR] = (unsigned short)(vreg[i] >> 16);
        }
        if (t + 1 < nTiles) {   // stage tile t+1
            uint64_t ga = (uint64_t)(uintptr_t)(Kld + (size_t)(kv0 + 64 + ldR) * HD);
            uint32_t la = lds_off32(&lK[buf ^ 1][ldR][ldSeg]);
            async_b128(la, ga); async_b128_o16(la, ga);
            const uint4* vp = (const uint4*)(Vld + (size_t)(kv0 + 64 + ldR) * HD);
            *(uint4*)&vreg[0] = vp[0];
            *(uint4*)&vreg[4] = vp[1];
            wait_async2();
        } else {
            wait_async0();
        }
        __syncthreads();

        // ---- S = Q K^T : four 16-wide column tiles, hd = 2 x 32 ----
        v8f sc[4] = {};
        #pragma unroll
        for (int nt = 0; nt < 4; ++nt) {
            #pragma unroll
            for (int ks = 0; ks < 2; ++ks) {
                v16bf kb = *(const v16bf*)(&lK[buf][nt*16 + mrow][ks*32 + 16*half]);
                sc[nt] = wmma_bf16(qa[ks], kb, sc[nt]);
            }
        }

        // ---- scale + causal mask + online softmax ----
        #pragma unroll
        for (int r = 0; r < 8; ++r) {
            int gm = qBase + wave*16 + r + 8*half;
            float mx = -3.0e38f;
            #pragma unroll
            for (int nt = 0; nt < 4; ++nt) {
                int gn = kv0 + nt*16 + mrow;
                float v = sc[nt][r] * scale;
                v = (gn <= gm) ? v : -3.0e38f;
                sc[nt][r] = v;
                mx = fmaxf(mx, v);
            }
            mx = red_max16(mx);
            float mnew  = fmaxf(mi[r], mx);
            float alpha = __expf(mi[r] - mnew);
            mi[r] = mnew;
            li[r] *= alpha;
            #pragma unroll
            for (int f = 0; f < 4; ++f) o[f][r] *= alpha;
            float psum = 0.0f;
            #pragma unroll
            for (int nt = 0; nt < 4; ++nt) {
                float pv = __expf(sc[nt][r] - mnew);   // masked -> exp(-huge)=0
                sc[nt][r] = pv;
                psum += pv;
            }
            li[r] += red_sum16(psum);
        }

        // ---- P (C-layout) -> LDS -> A-layout (wave-private) ----
        #pragma unroll
        for (int nt = 0; nt < 4; ++nt)
            #pragma unroll
            for (int r = 0; r < 8; ++r)
                lP[wave][r + 8*half][nt*16 + mrow] = f2bf(sc[nt][r]);
        asm volatile("s_wait_dscnt 0" ::: "memory");

        // ---- O += P @ V : kv = 2 x 32, hd = 4 x 16 ----
        #pragma unroll
        for (int ks = 0; ks < 2; ++ks) {
            v16bf pa = load_afrag(&lP[wave][mrow][ks*32], half);
            #pragma unroll
            for (int f = 0; f < 4; ++f) {
                v16bf vb = *(const v16bf*)(&lV[buf][f*16 + mrow][ks*32 + 16*half]);
                o[f] = wmma_bf16(pa, vb, o[f]);
            }
        }
        __syncthreads();
    }

    // ---- epilogue: O * (1/l) -> Ctx [B][S][768] bf16 ----
    const int b = bh / HEADS;
    const int h = bh % HEADS;
    #pragma unroll
    for (int r = 0; r < 8; ++r) {
        float inv = __builtin_amdgcn_rcpf(li[r]);
        int s = qBase + wave*16 + r + 8*half;
        #pragma unroll
        for (int f = 0; f < 4; ++f)
            Ctx[((size_t)b * SEQ + s) * DMODEL + h*HD + f*16 + mrow] = f2bf(o[f][r] * inv);
    }
}

// ===========================================================================
// Kernel 3: out = Ctx @ Wo^T + bo (f32 out). Fully async-staged, double-buffered.
// ===========================================================================
__global__ __launch_bounds__(256)
void out_proj_kernel(const unsigned short* __restrict__ Ctx,
                     const unsigned short* __restrict__ Wo,  // bf16
                     const float* __restrict__ bo,
                     float* __restrict__ Out)
{
    __shared__ __attribute__((aligned(32))) unsigned short lA[2][128][80];
    __shared__ __attribute__((aligned(32))) unsigned short lB[2][64][80];

    const int tid   = threadIdx.x;
    const int lane  = tid & 31;
    const int wave  = tid >> 5;
    const int waveM = wave & 3;
    const int waveN = wave >> 2;
    const int blockM = (blockIdx.x & 31) * 128;
    const int blockN = (blockIdx.x >> 5) * 64;
    const int mrow = lane & 15;
    const int half = lane >> 4;

    const int aR = tid >> 1, aK = (tid & 1) * 32;
    const int bN = tid >> 2, bK = (tid & 3) * 16;
    const unsigned short* Ald = Ctx + (size_t)(blockM + aR) * DMODEL + aK;
    const unsigned short* Bld = Wo  + (size_t)(blockN + bN) * DMODEL + bK;

    v8f acc[2][2] = {};
    const int nK = DMODEL / 64;

    {
        uint64_t ga = (uint64_t)(uintptr_t)Ald;
        uint32_t la = lds_off32(&lA[0][aR][aK]);
        async_b128(la, ga); async_b128_o16(la, ga);
        async_b128_o32(la, ga); async_b128_o48(la, ga);
        uint64_t gb = (uint64_t)(uintptr_t)Bld;
        uint32_t lb = lds_off32(&lB[0][bN][bK]);
        async_b128(lb, gb); async_b128_o16(lb, gb);
    }

    #pragma unroll 1    // keep single body so accumulators coalesce in-place
    for (int t = 0; t < nK; ++t) {
        const int buf = t & 1;
        if (t + 1 < nK) {
            int k0n = (t + 1) * 64;
            uint64_t ga = (uint64_t)(uintptr_t)(Ald + k0n);
            uint32_t la = lds_off32(&lA[buf ^ 1][aR][aK]);
            async_b128(la, ga); async_b128_o16(la, ga);
            async_b128_o32(la, ga); async_b128_o48(la, ga);
            uint64_t gb = (uint64_t)(uintptr_t)(Bld + k0n);
            uint32_t lb = lds_off32(&lB[buf ^ 1][bN][bK]);
            async_b128(lb, gb); async_b128_o16(lb, gb);
            wait_async6();
        } else {
            wait_async0();
        }
        __syncthreads();
        #pragma unroll
        for (int ks = 0; ks < 2; ++ks) {
            v16bf a0 = load_afrag(&lA[buf][waveM*32      + mrow][ks*32], half);
            v16bf a1 = load_afrag(&lA[buf][waveM*32 + 16 + mrow][ks*32], half);
            v16bf b0 = load_bfrag(&lB[buf][waveN*32      + mrow][ks*32], half);
            v16bf b1 = load_bfrag(&lB[buf][waveN*32 + 16 + mrow][ks*32], half);
            acc[0][0] = wmma_bf16(a0, b0, acc[0][0]);
            acc[0][1] = wmma_bf16(a0, b1, acc[0][1]);
            acc[1][0] = wmma_bf16(a1, b0, acc[1][0]);
            acc[1][1] = wmma_bf16(a1, b1, acc[1][1]);
        }
        __syncthreads();
    }

    #pragma unroll
    for (int mi = 0; mi < 2; ++mi)
    #pragma unroll
    for (int ni = 0; ni < 2; ++ni) {
        int gn = blockN + waveN*32 + ni*16 + mrow;
        float bb = bo[gn];
        #pragma unroll
        for (int r = 0; r < 8; ++r) {
            int gm = blockM + waveM*32 + mi*16 + r + 8*half;
            Out[(size_t)gm * DMODEL + gn] = acc[mi][ni][r] + bb;
        }
    }
}

// ===========================================================================
extern "C" void kernel_launch(void* const* d_in, const int* in_sizes, int n_in,
                              void* d_out, int out_size, void* d_ws, size_t ws_size,
                              hipStream_t stream) {
    const float* x  = (const float*)d_in[0];
    const float* Wq = (const float*)d_in[1];
    const float* bq = (const float*)d_in[2];
    const float* Wk = (const float*)d_in[3];
    const float* bk = (const float*)d_in[4];
    const float* Wv = (const float*)d_in[5];
    const float* bv = (const float*)d_in[6];
    const float* Wo = (const float*)d_in[7];
    const float* bo = (const float*)d_in[8];
    float* out = (float*)d_out;

    const size_t HSZ = (size_t)2 * HEADS * SEQ * HD;      // 3,145,728
    const size_t XSZ = (size_t)2 * SEQ * DMODEL;          // 3,145,728
    const size_t WSZ = (size_t)DMODEL * DMODEL;           //   589,824
    unsigned short* Qb  = (unsigned short*)d_ws;          // total ~35 MB
    unsigned short* Kb  = Qb  + HSZ;
    unsigned short* Vb  = Kb  + HSZ;
    unsigned short* Cx  = Vb  + HSZ;
    unsigned short* Xb  = Cx  + HSZ;
    unsigned short* Wqb = Xb  + XSZ;
    unsigned short* Wkb = Wqb + WSZ;
    unsigned short* Wvb = Wkb + WSZ;
    unsigned short* Wob = Wvb + WSZ;

    dim3 block(256);
    // one-shot f32 -> bf16 conversion (done once; GEMM loops become pure DMA)
    cvt_f32_bf16_kernel<<<dim3((int)(XSZ / 8 / 256)), block, 0, stream>>>(x,  Xb,  (int)(XSZ / 8));
    cvt_f32_bf16_kernel<<<dim3((int)(WSZ / 8 / 256)), block, 0, stream>>>(Wq, Wqb, (int)(WSZ / 8));
    cvt_f32_bf16_kernel<<<dim3((int)(WSZ / 8 / 256)), block, 0, stream>>>(Wk, Wkb, (int)(WSZ / 8));
    cvt_f32_bf16_kernel<<<dim3((int)(WSZ / 8 / 256)), block, 0, stream>>>(Wv, Wvb, (int)(WSZ / 8));
    cvt_f32_bf16_kernel<<<dim3((int)(WSZ / 8 / 256)), block, 0, stream>>>(Wo, Wob, (int)(WSZ / 8));

    dim3 grid(384);
    proj_rope_kernel<<<grid, block, 0, stream>>>(Xb, Wqb, bq, Qb, 1);
    proj_rope_kernel<<<grid, block, 0, stream>>>(Xb, Wkb, bk, Kb, 1);
    proj_rope_kernel<<<grid, block, 0, stream>>>(Xb, Wvb, bv, Vb, 0);
    flash_attn_kernel<<<grid, block, 0, stream>>>(Qb, Kb, Vb, Cx);
    out_proj_kernel<<<grid, block, 0, stream>>>(Cx, Wob, bo, out);
}
